// RNN_6975026889164
// MI455X (gfx1250) — compile-verified, matrix operand order checked by reading, not run on hardware
//
#include <hip/hip_runtime.h>

// ---------------------------------------------------------------------------
// CDNA5 (gfx1250) bidirectional 2-layer tanh-RNN.
//   B=64, S=2048, I=256, H=256.
//   1) x -> bf16 with (B,S,I)->(S,B,I) transpose
//   2) weights -> bf16
//   3) pre = x.Wx^T + bias via v_wmma_f32_16x16x32_bf16, stored in
//      C-fragment order so the scan reloads it with two b128 per lane
//   4) persistent 2-block scan: h = tanh(pre + h.Wh^T); Wh B-frags pinned
//      in registers, h in padded LDS, next-step pre prefetched, fast tanh
//   5) repeat for layer 1 writing d_out (B,S,2H) f32; zero hn.
// ---------------------------------------------------------------------------

typedef __bf16 bf16_t;
typedef __attribute__((ext_vector_type(16))) __bf16 v16bf;
typedef __attribute__((ext_vector_type(8)))  float  v8f;
typedef __attribute__((ext_vector_type(4)))  float  f32x4;
typedef __attribute__((ext_vector_type(4)))  unsigned int u32x4;

constexpr int Bsz = 64;
constexpr int Seq = 2048;
constexpr int Iin = 256;
constexpr int Hid = 256;

union FragBF { v16bf v; u32x4 q[2]; };
union FragF  { v8f   v; f32x4 q[2]; };

// 16x32 bf16 WMMA A-fragment loader (also B-fragment when the source is the
// N x K row-major weight matrix).  ISA 7.12.2: lanes 0-15 row=lane,
// K in {0..7,16..23}; lanes 16-31 K in {8..15,24..31}.  Two b128 per lane.
__device__ __forceinline__ v16bf load_frag(const bf16_t* __restrict__ tile,
                                           int ld, int k0, int r15, int hi8) {
    const bf16_t* p = tile + (size_t)r15 * ld + k0 + hi8;
    FragBF f;
    f.q[0] = *(const u32x4*)(p);
    f.q[1] = *(const u32x4*)(p + 16);
    return f.v;
}

// tanh via hw v_exp_f32: 1 - 2/(e^{2x}+1).  Saturates correctly at +-inf.
__device__ __forceinline__ float fast_tanh(float x) {
    const float e = __expf(2.0f * x);
    return 1.0f - __fdividef(2.0f, e + 1.0f);
}

// --------------------------- data conversion -------------------------------

__global__ void xpose_bf16_kernel(const float* __restrict__ x,
                                  bf16_t* __restrict__ xbf) {
    const int i  = threadIdx.x;
    const int sb = blockIdx.x;          // s*B + b
    const int s  = sb >> 6;
    const int b  = sb & 63;
    xbf[(size_t)sb * Iin + i] = (bf16_t)x[((size_t)b * Seq + s) * Iin + i];
}

__global__ void cvt_bf16_kernel(const float* __restrict__ in,
                                bf16_t* __restrict__ out, int n) {
    int i = blockIdx.x * blockDim.x + threadIdx.x;
    if (i < n) out[i] = (bf16_t)in[i];
}

__global__ void zero_f32_kernel(float* __restrict__ p, int n) {
    int i = blockIdx.x * blockDim.x + threadIdx.x;
    if (i < n) p[i] = 0.f;
}

// --------------------------- pre-activation GEMM ---------------------------
// pre[(s,tt,nt,lane)*8 + r] = fragment-ordered (x.Wx^T + bias).
// Block = 256 threads = 8 waves; wave -> 16(M) x 64(N) strip (4 C tiles).
// M-tiles never straddle s (64 rows per s, tiles of 16).
__global__ void __launch_bounds__(256)
pre_gemm_kernel(const bf16_t* __restrict__ A, int K,
                const bf16_t* __restrict__ Wd0, const bf16_t* __restrict__ Wd1,
                int Wld,
                const float* __restrict__ bias0, const float* __restrict__ bias1,
                float* __restrict__ pre0, float* __restrict__ pre1) {
    const bf16_t* W    = blockIdx.z ? Wd1   : Wd0;
    const float*  bias = blockIdx.z ? bias1 : bias0;
    float*        pre  = blockIdx.z ? pre1  : pre0;

    const int lane = threadIdx.x & 31;
    const int wave = threadIdx.x >> 5;
    const int r15  = lane & 15;
    const int hi8  = (lane & 16) >> 1;          // 0 or 8
    const int m0     = (blockIdx.x * 8 + wave) * 16;
    const int n_base = blockIdx.y * 64;

    v8f acc[4];
#pragma unroll
    for (int t = 0; t < 4; ++t) {
        const float bv = bias[n_base + t * 16 + r15];
#pragma unroll
        for (int r = 0; r < 8; ++r) acc[t][r] = bv;
    }

    const bf16_t* Arow = A + (size_t)m0 * K;
    for (int k0 = 0; k0 < K; k0 += 32) {
        const v16bf a = load_frag(Arow, K, k0, r15, hi8);
#pragma unroll
        for (int t = 0; t < 4; ++t) {
            const v16bf b = load_frag(W + (size_t)(n_base + t * 16) * Wld,
                                      Wld, k0, r15, hi8);
            acc[t] = __builtin_amdgcn_wmma_f32_16x16x32_bf16(
                false, a, false, b, (short)0, acc[t], false, false);
        }
    }

    // Fragment-ordered store: two b128 per lane per tile, fully coalesced.
    const int s_  = m0 >> 6;
    const int tt_ = (m0 >> 4) & 3;
#pragma unroll
    for (int t = 0; t < 4; ++t) {
        const int ntg = (n_base >> 4) + t;
        float* dst = pre +
            ((((size_t)s_ * 4 + tt_) * 16 + ntg) * 32 + lane) * 8;
        FragF u; u.v = acc[t];
        *(f32x4*)(dst)     = u.q[0];
        *(f32x4*)(dst + 4) = u.q[1];
    }
}

// ------------------------------- recurrence --------------------------------
// Persistent scan: grid = 2 blocks (dir), 512 threads = 16 waves.
// Wave w owns N-tile w and all 4 M-tiles; its 8 Wh B-frags live in registers
// for the whole scan.  h lives in LDS with row stride 264 bf16 (conflict-free
// b128 reads).  Next step's pre is prefetched behind the tanh + barriers.
__global__ void __launch_bounds__(512)
rnn_scan_kernel(const float* __restrict__ pre_f, const float* __restrict__ pre_b,
                const bf16_t* __restrict__ Wh_f, const bf16_t* __restrict__ Wh_b,
                int WhLd,
                bf16_t* __restrict__ ybf,   // layer0: (S,B,2H) bf16, else null
                float*  __restrict__ yout)  // layer1: (B,S,2H) f32,  else null
{
    const int dir = blockIdx.x;
    const float*  pre = dir ? pre_b : pre_f;
    const bf16_t* Wh  = dir ? Wh_b  : Wh_f;

    const int lane = threadIdx.x & 31;
    const int wave = threadIdx.x >> 5;   // 0..15 == N-tile index
    const int r15  = lane & 15;
    const int hi8  = (lane & 16) >> 1;
    const int nt   = wave;

    constexpr int LDH = Hid + 8;                       // 264 bf16 / row
    __shared__ bf16_t hbuf[Bsz * LDH];                 // 33792 bytes

    // Pin this wave's 8 Wh B-fragments in registers.
    v16bf breg[8];
#pragma unroll
    for (int kk = 0; kk < 8; ++kk)
        breg[kk] = load_frag(Wh + (size_t)(nt * 16) * WhLd, WhLd,
                             kk * 32, r15, hi8);

    // h0 = 0
    for (int idx = threadIdx.x; idx < Bsz * LDH; idx += 512)
        hbuf[idx] = (bf16_t)0.f;
    __syncthreads();

    const bool is_l0 = (ybf != nullptr);

    // Fragment-ordered pre loader: two b128 per lane per tile.
    auto load_pre = [&](int s, v8f* pv) {
#pragma unroll
        for (int tt = 0; tt < 4; ++tt) {
            const float* p = pre +
                ((((size_t)s * 4 + tt) * 16 + nt) * 32 + lane) * 8;
            FragF u;
            u.q[0] = *(const f32x4*)(p);
            u.q[1] = *(const f32x4*)(p + 4);
            pv[tt] = u.v;
        }
    };

    v8f pv[4];
    load_pre(dir ? (Seq - 1) : 0, pv);

    for (int t = 0; t < Seq; ++t) {
        const int s = dir ? (Seq - 1 - t) : t;

        // GEMM phase: c[tt] = pre + h.Wh^T
        v8f c[4];
#pragma unroll
        for (int tt = 0; tt < 4; ++tt) {
            v8f acc = pv[tt];
#pragma unroll
            for (int kk = 0; kk < 8; ++kk) {
                const v16bf a = load_frag(hbuf + (tt * 16) * LDH, LDH,
                                          kk * 32, r15, hi8);
                acc = __builtin_amdgcn_wmma_f32_16x16x32_bf16(
                    false, a, false, breg[kk], (short)0, acc, false, false);
            }
            c[tt] = acc;
        }

        // Prefetch next step's pre while tanh + barriers + stores run.
        if (t + 1 < Seq) load_pre(dir ? (Seq - 2 - t) : (t + 1), pv);

#pragma unroll
        for (int tt = 0; tt < 4; ++tt)
#pragma unroll
            for (int r = 0; r < 8; ++r) c[tt][r] = fast_tanh(c[tt][r]);

        __syncthreads();   // everyone done reading old h
#pragma unroll
        for (int tt = 0; tt < 4; ++tt) {
#pragma unroll
            for (int r = 0; r < 8; ++r) {
                const int m = tt * 16 + hi8 + r;
                const int n = nt * 16 + r15;
                const float hv = c[tt][r];
                hbuf[m * LDH + n] = (bf16_t)hv;
                if (is_l0)
                    ybf[((size_t)s * Bsz + m) * (2 * Hid) + dir * Hid + n] =
                        (bf16_t)hv;
                else
                    yout[((size_t)m * Seq + s) * (2 * Hid) + dir * Hid + n] = hv;
            }
        }
        __syncthreads();   // new h visible before next step's reads
    }
}

// ------------------------------- launcher ----------------------------------

extern "C" void kernel_launch(void* const* d_in, const int* in_sizes, int n_in,
                              void* d_out, int out_size, void* d_ws,
                              size_t ws_size, hipStream_t stream) {
    (void)in_sizes; (void)n_in; (void)out_size; (void)ws_size;

    const float* x   = (const float*)d_in[0];
    const float* W0f = (const float*)d_in[1];
    const float* b0f = (const float*)d_in[2];
    const float* W0b = (const float*)d_in[3];
    const float* b0b = (const float*)d_in[4];
    const float* W1f = (const float*)d_in[5];
    const float* b1f = (const float*)d_in[6];
    const float* W1b = (const float*)d_in[7];
    const float* b1b = (const float*)d_in[8];
    float* out = (float*)d_out;

    // Workspace (~403 MB).  y0 aliases the xbf region: xbf is dead after the
    // layer-0 pre-GEMM; y0 is written only afterwards (stream-ordered).
    char* ws = (char*)d_ws;
    const size_t M = (size_t)Seq * Bsz;                 // 131072 rows
    bf16_t* y0   = (bf16_t*)ws;                          // (S,B,2H) bf16
    bf16_t* xbf  = (bf16_t*)ws;                          // (S,B,I)  bf16 (alias)
    size_t off   = M * 2 * Hid * sizeof(bf16_t);
    bf16_t* W0fb = (bf16_t*)(ws + off); off += (size_t)Hid * 512 * 2;
    bf16_t* W0bb = (bf16_t*)(ws + off); off += (size_t)Hid * 512 * 2;
    bf16_t* W1fb = (bf16_t*)(ws + off); off += (size_t)Hid * 768 * 2;
    bf16_t* W1bb = (bf16_t*)(ws + off); off += (size_t)Hid * 768 * 2;
    off = (off + 255) & ~(size_t)255;
    float* preF  = (float*)(ws + off);  off += M * Hid * sizeof(float);
    float* preB  = (float*)(ws + off);  off += M * Hid * sizeof(float);

    // 1) x -> bf16, (B,S,I) -> (S,B,I)
    xpose_bf16_kernel<<<Seq * Bsz, Iin, 0, stream>>>(x, xbf);

    // 2) weights -> bf16
    cvt_bf16_kernel<<<(Hid * 512 + 255) / 256, 256, 0, stream>>>(W0f, W0fb, Hid * 512);
    cvt_bf16_kernel<<<(Hid * 512 + 255) / 256, 256, 0, stream>>>(W0b, W0bb, Hid * 512);
    cvt_bf16_kernel<<<(Hid * 768 + 255) / 256, 256, 0, stream>>>(W1f, W1fb, Hid * 768);
    cvt_bf16_kernel<<<(Hid * 768 + 255) / 256, 256, 0, stream>>>(W1b, W1bb, Hid * 768);

    const dim3 gemm_grid((unsigned)(M / 128), Hid / 64, 2);

    // 3) layer-0 pre-activations, both directions
    pre_gemm_kernel<<<gemm_grid, 256, 0, stream>>>(
        xbf, Iin, W0fb, W0bb, Iin + Hid, b0f, b0b, preF, preB);

    // 4) layer-0 scan (fwd + bwd persistent blocks) -> y0
    rnn_scan_kernel<<<2, 512, 0, stream>>>(
        preF, preB, W0fb + Iin, W0bb + Iin, Iin + Hid, y0, nullptr);

    // 5) layer-1 pre-activations (K = 2H = 512), pre buffers reused
    pre_gemm_kernel<<<gemm_grid, 256, 0, stream>>>(
        y0, 2 * Hid, W1fb, W1bb, 3 * Hid, b1f, b1b, preF, preB);

    // 6) layer-1 scan -> d_out (B,S,2H) f32
    rnn_scan_kernel<<<2, 512, 0, stream>>>(
        preF, preB, W1fb + 2 * Hid, W1bb + 2 * Hid, 3 * Hid, nullptr, out);

    // 7) hn = zeros(4,B,H)
    const int hn_n = 4 * Bsz * Hid;
    zero_f32_kernel<<<(hn_n + 255) / 256, 256, 0, stream>>>(
        out + (size_t)Bsz * Seq * 2 * Hid, hn_n);
}
